// MultiModalGraphTransformer_86105504350513
// MI455X (gfx1250) — compile-verified, matrix operand order checked by reading, not run on hardware
//
#include <hip/hip_runtime.h>
#include <hip/hip_bf16.h>
#include <hip/hip_fp16.h>

typedef __attribute__((ext_vector_type(16))) _Float16 v16h;
typedef __attribute__((ext_vector_type(8)))  float    v8f;

#define HD    64   // hidden dim
#define NHEAD 4
#define CH    16   // channels per head

// ---- monotone float <-> uint encoding for atomicMax on floats ----
__device__ __forceinline__ unsigned f32_ord(float f) {
  unsigned u = __float_as_uint(f);
  return (u & 0x80000000u) ? ~u : (u | 0x80000000u);
}
__device__ __forceinline__ float ord_f32(unsigned u) {
  return (u & 0x80000000u) ? __uint_as_float(u ^ 0x80000000u)
                           : __uint_as_float(~u);
}

// ---- zero the accumulators (amax encoded-0 acts as -inf) ----
__global__ void init_kernel(unsigned* amax, float* denom, float* pooled,
                            float* cnts, int nh, int gp, int g) {
  int t = blockIdx.x * blockDim.x + threadIdx.x;
  if (t < nh) { amax[t] = 0u; denom[t] = 0.0f; }
  if (t < gp) pooled[t] = 0.0f;
  if (t < g)  cnts[t]   = 0.0f;
}

// ---- h = x @ W_fe.T + b_fe   (K = 2, pure VALU) ----
__global__ void feat_kernel(const float* __restrict__ x,
                            const float* __restrict__ Wfe,
                            const float* __restrict__ bfe,
                            float* __restrict__ h, int n) {
  int t = blockIdx.x * blockDim.x + threadIdx.x;
  if (t >= n * HD) return;
  int node = t >> 6, d = t & 63;
  h[t] = fmaf(x[node * 2], Wfe[d * 2],
         fmaf(x[node * 2 + 1], Wfe[d * 2 + 1], bfe[d]));
}

// ---- q/k/v/skip projections via WMMA f32_16x16x32_f16 ----
// block = 4 waves; wave w computes out_w[16 x 64] = h_tile @ W_w.T + b_w
__global__ void __launch_bounds__(128)
qkvs_kernel(const float* __restrict__ h,
            const float* __restrict__ Wq, const float* __restrict__ bq,
            const float* __restrict__ Wk, const float* __restrict__ bk,
            const float* __restrict__ Wv, const float* __restrict__ bv,
            const float* __restrict__ Ws, const float* __restrict__ bs,
            float* __restrict__ q, float* __restrict__ k,
            float* __restrict__ v, float* __restrict__ agg, int n) {
  __shared__ _Float16 hs[16][72];        // h tile, f16
  __shared__ _Float16 wsm[4][64][72];    // one 64x64 f16 weight matrix per wave

  const int wave = threadIdx.x >> 5;
  const int lane = threadIdx.x & 31;
  const int row0 = blockIdx.x * 16;

  const float* W;  const float* bias;  float* out;
  if      (wave == 0) { W = Wq; bias = bq; out = q;   }
  else if (wave == 1) { W = Wk; bias = bk; out = k;   }
  else if (wave == 2) { W = Wv; bias = bv; out = v;   }
  else                { W = Ws; bias = bs; out = agg; }

  // stage this wave's weight matrix (L2-resident) into LDS as f16
  for (int i = lane; i < 64 * 64; i += 32)
    wsm[wave][i >> 6][i & 63] = (_Float16)W[i];
  // stage the 16x64 h tile (whole block cooperates)
  for (int i = threadIdx.x; i < 16 * 64; i += 128) {
    int r = i >> 6, c = i & 63;
    int gr = row0 + r;
    hs[r][c] = (gr < n) ? (_Float16)h[(size_t)gr * HD + c] : (_Float16)0.0f;
  }
  __syncthreads();

  const int laneLo = lane & 15;
  const int laneHi = (lane >> 4) & 1;

  // A fragments (16x32 f16 layout): lane row = laneLo, K set offset +8 for hi half.
  // Per-lane elements are two contiguous 8xf16 runs -> lowers to ds_load_b128 pairs.
  v16h afrag[2];
#pragma unroll
  for (int kt = 0; kt < 2; ++kt) {
#pragma unroll
    for (int i = 0; i < 16; ++i) {
      int kk = kt * 32 + ((i >> 3) << 4) + (((i >> 1) & 3) << 1) + (i & 1) + laneHi * 8;
      afrag[kt][i] = hs[laneLo][kk];
    }
  }

  const bool fullTile = (row0 + 16 <= n);   // N % 16 == 0 -> always true at runtime

#pragma unroll
  for (int ct = 0; ct < 4; ++ct) {       // 4 column tiles of 16
    v8f acc = {};
    const int ncol = ct * 16 + laneLo;   // output column (== weight row, out = h @ W.T)
#pragma unroll
    for (int kt = 0; kt < 2; ++kt) {     // K = 64 -> 2 WMMA steps
      v16h bfrag;
#pragma unroll
      for (int i = 0; i < 16; ++i) {
        int kk = kt * 32 + ((i >> 3) << 4) + (((i >> 1) & 3) << 1) + (i & 1) + laneHi * 8;
        bfrag[i] = wsm[wave][ncol][kk];  // B[k][n] = W[n][k]
      }
      acc = __builtin_amdgcn_wmma_f32_16x16x32_f16(
          false, afrag[kt], false, bfrag, (short)0, acc, false, false);
    }
    const float bb = bias[ncol];
    // D layout: M = r + laneHi*8, N = laneLo
    float* orow = out + (size_t)(row0 + laneHi * 8) * HD + ncol;
    if (fullTile) {
#pragma unroll
      for (int r = 0; r < 8; ++r)
        orow[(size_t)r * HD] = acc[r] + bb;
    } else {
#pragma unroll
      for (int r = 0; r < 8; ++r)
        if (row0 + laneHi * 8 + r < n) orow[(size_t)r * HD] = acc[r] + bb;
    }
  }
}

// ---- pass A: alpha[e,h] = <q[dst], k[src]+ea> / sqrt(C); atomic segment max ----
__global__ void edge_alpha_kernel(const int* __restrict__ eidx,
                                  const float* __restrict__ eattr,
                                  const float* __restrict__ Wedge,
                                  const float* __restrict__ q,
                                  const float* __restrict__ k,
                                  float* __restrict__ alpha,
                                  unsigned* __restrict__ amax, int E) {
  int t = blockIdx.x * blockDim.x + threadIdx.x;
  if (t >= E * NHEAD) return;
  int e = t >> 2, hd = t & 3;
  int s = eidx[e], d = eidx[E + e];
  float ew = eattr[e];
  const float4* qp = (const float4*)(q + (size_t)d * HD + hd * CH);
  const float4* kp = (const float4*)(k + (size_t)s * HD + hd * CH);
  const float4* we = (const float4*)(Wedge + hd * CH);
  float acc = 0.0f;
#pragma unroll
  for (int c = 0; c < CH / 4; ++c) {
    float4 qq = qp[c], kk = kp[c], ww = we[c];
    acc = fmaf(qq.x, fmaf(ew, ww.x, kk.x), acc);
    acc = fmaf(qq.y, fmaf(ew, ww.y, kk.y), acc);
    acc = fmaf(qq.z, fmaf(ew, ww.z, kk.z), acc);
    acc = fmaf(qq.w, fmaf(ew, ww.w, kk.w), acc);
  }
  acc *= 0.25f;                       // 1/sqrt(16)
  alpha[t] = acc;
  atomicMax(&amax[(size_t)d * NHEAD + hd], f32_ord(acc));
}

// ---- pass B: ae = exp(alpha - amax[dst]); denom[dst] += ae (overwrite alpha) ----
__global__ void edge_expsum_kernel(const int* __restrict__ eidx,
                                   float* __restrict__ alpha,
                                   const unsigned* __restrict__ amax,
                                   float* __restrict__ denom, int E) {
  int t = blockIdx.x * blockDim.x + threadIdx.x;
  if (t >= E * NHEAD) return;
  int e = t >> 2, hd = t & 3;
  int d = eidx[E + e];
  float m  = ord_f32(amax[(size_t)d * NHEAD + hd]);
  float ae = __expf(alpha[t] - m);
  alpha[t] = ae;
  atomicAdd(&denom[(size_t)d * NHEAD + hd], ae);
}

// ---- pass C: agg[dst] += (ae/denom) * (v[src] + ea) ----
__global__ void edge_msg_kernel(const int* __restrict__ eidx,
                                const float* __restrict__ eattr,
                                const float* __restrict__ Wedge,
                                const float* __restrict__ v,
                                const float* __restrict__ alpha,
                                const float* __restrict__ denom,
                                float* __restrict__ agg, int E) {
  int t = blockIdx.x * blockDim.x + threadIdx.x;
  if (t >= E * NHEAD) return;
  int e = t >> 2, hd = t & 3;
  int s = eidx[e], d = eidx[E + e];
  float attn = alpha[t] / fmaxf(denom[(size_t)d * NHEAD + hd], 1e-16f);
  float ew = eattr[e];
  const float4* vp = (const float4*)(v + (size_t)s * HD + hd * CH);
  const float4* we = (const float4*)(Wedge + hd * CH);
  float* ap = agg + (size_t)d * HD + hd * CH;
#pragma unroll
  for (int c = 0; c < CH / 4; ++c) {
    float4 vv = vp[c], ww = we[c];
    atomicAdd(&ap[c * 4 + 0], attn * fmaf(ew, ww.x, vv.x));
    atomicAdd(&ap[c * 4 + 1], attn * fmaf(ew, ww.y, vv.y));
    atomicAdd(&ap[c * 4 + 2], attn * fmaf(ew, ww.z, vv.z));
    atomicAdd(&ap[c * 4 + 3], attn * fmaf(ew, ww.w, vv.w));
  }
}

// ---- ReLU (agg already contains skip) + mean-pool accumulation ----
__global__ void pool_kernel(const float* __restrict__ agg,
                            const int* __restrict__ batch,
                            float* __restrict__ pooled,
                            float* __restrict__ cnts, int n) {
  int t = blockIdx.x * blockDim.x + threadIdx.x;
  if (t >= n * HD) return;
  int node = t >> 6, d = t & 63;
  int g = batch[node];
  atomicAdd(&pooled[(size_t)g * HD + d], fmaxf(agg[t], 0.0f));
  if (d == 0) atomicAdd(&cnts[g], 1.0f);
}

// ---- tiny classifier head: one thread per graph ----
__global__ void cls_kernel(const float* __restrict__ pooled,
                           const float* __restrict__ cnts,
                           const float* __restrict__ Wc1, const float* __restrict__ bc1,
                           const float* __restrict__ Wc2, const float* __restrict__ bc2,
                           float* __restrict__ out, int G_, int NC_) {
  int g = blockIdx.x * blockDim.x + threadIdx.x;
  if (g >= G_) return;
  float inv = 1.0f / fmaxf(cnts[g], 1.0f);
  float p[HD];
#pragma unroll
  for (int d = 0; d < HD; ++d) p[d] = pooled[(size_t)g * HD + d] * inv;
  float z[HD / 2];
  for (int j = 0; j < HD / 2; ++j) {
    float s = bc1[j];
#pragma unroll
    for (int d = 0; d < HD; ++d) s = fmaf(p[d], Wc1[j * HD + d], s);
    z[j] = fmaxf(s, 0.0f);
  }
  for (int o = 0; o < NC_; ++o) {
    float s = bc2[o];
#pragma unroll
    for (int j = 0; j < HD / 2; ++j) s = fmaf(z[j], Wc2[o * (HD / 2) + j], s);
    out[g * NC_ + o] = s;
  }
}

extern "C" void kernel_launch(void* const* d_in, const int* in_sizes, int n_in,
                              void* d_out, int out_size, void* d_ws, size_t ws_size,
                              hipStream_t stream) {
  const float* x     = (const float*)d_in[0];
  const int*   eidx  = (const int*)  d_in[1];
  const float* eattr = (const float*)d_in[2];
  const int*   batch = (const int*)  d_in[3];
  const float* Wfe   = (const float*)d_in[4];
  const float* bfe   = (const float*)d_in[5];
  const float* Wedge = (const float*)d_in[6];
  const float* Wq = (const float*)d_in[7];  const float* bq = (const float*)d_in[8];
  const float* Wk = (const float*)d_in[9];  const float* bk = (const float*)d_in[10];
  const float* Wv = (const float*)d_in[11]; const float* bv = (const float*)d_in[12];
  const float* Ws = (const float*)d_in[13]; const float* bs = (const float*)d_in[14];
  const float* Wc1 = (const float*)d_in[15]; const float* bc1 = (const float*)d_in[16];
  const float* Wc2 = (const float*)d_in[17]; const float* bc2 = (const float*)d_in[18];

  const int n   = in_sizes[0] / 2;   // x is [N, 2]
  const int E   = in_sizes[2];       // edge_attr is [E, 1]
  const int NC_ = in_sizes[18];      // bc2 length
  const int G_  = out_size / NC_;

  // workspace carve-up (floats)
  const size_t N64 = (size_t)n * HD;
  const size_t EH  = (size_t)E * NHEAD;
  const size_t NH  = (size_t)n * NHEAD;
  float* wsf    = (float*)d_ws;
  float* h      = wsf;
  float* q      = h + N64;
  float* k      = q + N64;
  float* v      = k + N64;
  float* agg    = v + N64;           // initialized with skip projection by qkvs_kernel
  float* alpha  = agg + N64;         // [E, H]; reused as ae
  unsigned* amax = (unsigned*)(alpha + EH);  // [N, H] encoded
  float* denom  = (float*)(amax + NH);       // [N, H]
  float* pooled = denom + NH;                // [G, 64]
  float* cnts   = pooled + (size_t)G_ * HD;  // [G]

  const int B = 256;
  const int nh = (int)NH, eh = (int)EH, nd = (int)N64;

  init_kernel<<<(nh + B - 1) / B, B, 0, stream>>>(amax, denom, pooled, cnts,
                                                  nh, G_ * HD, G_);
  feat_kernel<<<(nd + B - 1) / B, B, 0, stream>>>(x, Wfe, bfe, h, n);
  qkvs_kernel<<<(n + 15) / 16, 128, 0, stream>>>(h, Wq, bq, Wk, bk, Wv, bv,
                                                 Ws, bs, q, k, v, agg, n);
  edge_alpha_kernel<<<(eh + B - 1) / B, B, 0, stream>>>(eidx, eattr, Wedge, q, k,
                                                        alpha, amax, E);
  edge_expsum_kernel<<<(eh + B - 1) / B, B, 0, stream>>>(eidx, alpha, amax, denom, E);
  edge_msg_kernel<<<(eh + B - 1) / B, B, 0, stream>>>(eidx, eattr, Wedge, v,
                                                      alpha, denom, agg, E);
  pool_kernel<<<(nd + B - 1) / B, B, 0, stream>>>(agg, batch, pooled, cnts, n);
  cls_kernel<<<(G_ + 63) / 64, 64, 0, stream>>>(pooled, cnts, Wc1, bc1, Wc2, bc2,
                                                (float*)d_out, G_, NC_);
}
